// Euler_Lagrange_Predictor_82892868813315
// MI455X (gfx1250) — matrix operand order repaired and checked by reading, not compiled, not run on hardware
//
#include <hip/hip_runtime.h>
#include <hip/hip_bf16.h>

typedef __attribute__((ext_vector_type(16))) _Float16 v16h;
typedef __attribute__((ext_vector_type(8)))  float    v8f;
typedef __attribute__((ext_vector_type(4)))  float    v4f;

union HalfFrag { v16h v; v4f q[2]; };

#if defined(__has_builtin)
#if __has_builtin(__builtin_amdgcn_sched_barrier)
#define SCHED_FENCE() __builtin_amdgcn_sched_barrier(0)
#endif
#endif
#ifndef SCHED_FENCE
#define SCHED_FENCE() asm volatile("" ::: "memory")
#endif

// Problem constants
#define NB   8
#define NT   12
#define NH   128
#define NW   128
#define HW   16384       // NH*NW
#define HID  64
#define NG   256         // 4*HID gate channels
#define KREAL 585        // (1 + HID)*9
#define KP    608        // padded to 19*32
#define NKC   19         // KP/32
#define AS    40         // LDS stride (halfs) for A tile rows
#define BS    40         // LDS stride (halfs) for B tile rows

// ---------------------------------------------------------------------------
// Pack Wx (4,HID,1,3,3) + Wh (4,HID,HID,3,3) into im2col-ordered f16 [NG][KP]
// ---------------------------------------------------------------------------
__global__ void pack_weights(const float* __restrict__ Wx,
                             const float* __restrict__ Wh,
                             _Float16* __restrict__ Wp) {
    int idx = blockIdx.x * blockDim.x + threadIdx.x;
    if (idx >= NG * KP) return;
    int o = idx / KP, kk = idx - o * KP;
    float v = 0.f;
    if (kk < KREAL) {
        int c = kk / 9, r = kk - c * 9;
        int g = o >> 6, u = o & 63;
        if (c == 0) v = Wx[(g * HID + u) * 9 + r];
        else        v = Wh[((g * HID + u) * HID + (c - 1)) * 9 + r];
    }
    Wp[idx] = (_Float16)v;
}

// ---------------------------------------------------------------------------
// Fused gate GEMM: G[S][o] = sum_k im2col([xin;h])[S][k] * Wp[o][k] + bias[o]
// M = NB*HW pixels, N = 256, K = 608. f16 WMMA, f32 accumulate.
// Block: 512 threads (16 wave32). BM = 128 pixels (one image row), BN = 256.
// Wave tile: 2 m-tiles x 4 n-tiles = 8 WMMAs/chunk.
// Double-buffered LDS; pipeline order pinned with sched fences:
//   global loads | FENCE | frag loads + WMMAs | FENCE | lds stores | barrier
// ---------------------------------------------------------------------------
__global__ __launch_bounds__(512)
void gates_wmma(const float* __restrict__ xin, int xinBStride,
                const float* __restrict__ hbuf,
                const _Float16* __restrict__ Wp,
                const float* __restrict__ bias,
                float* __restrict__ G) {
    __shared__ __align__(16) _Float16 As[2][128 * AS];
    __shared__ __align__(16) _Float16 Bs[2][256 * BS];

    const int tid  = threadIdx.x;
    const int wave = tid >> 5;
    const int lane = tid & 31;
    const int mtA  = (wave & 3) * 2;  // first of 2 m-tiles (0..7)
    const int ngrp = wave >> 2;       // 0..3 : 64-channel group
    const int lrow = lane & 15;
    const int lhi  = lane >> 4;       // 0: lanes 0-15, 1: lanes 16-31
    const int S0   = blockIdx.x * 128;    // one full image row
    const int bimg = blockIdx.x >> 7;
    const int y    = blockIdx.x & 127;

    // A-staging role: k fixed per thread, m = mbase + 16*j (j<8)
    const int ka    = tid & 31;
    const int mbase = tid >> 5;       // 0..15
    // B-staging role: 256 rows x 32 k, 2 threads per row
    const int bn  = tid & 255;
    const int bhs = tid >> 8;         // 0/1 -> which 16-half chunk

    v8f acc[8] = {v8f{}, v8f{}, v8f{}, v8f{}, v8f{}, v8f{}, v8f{}, v8f{}};

    float av[8], af_[8];
    v4f   bv0, bv1;

    // Global -> regs for K-chunk kc (branchless, clamped addresses)
    auto loadA = [&](int kc) {
        int kk = kc + ka;
        int c  = kk / 9, r = kk - c * 9;
        int ry = r / 3;
        int dy = ry - 1, dx = r - ry * 3 - 1;
        int yy = y + dy;
        int rowok = (kk < KREAL) & ((unsigned)yy < (unsigned)NH);
        int yyc = min(max(yy, 0), NH - 1);
        int cc  = min(c, HID);                 // clamp padded-K channel
        const float* plane = (cc == 0)
            ? (xin + (size_t)bimg * xinBStride)
            : (hbuf + ((size_t)(bimg * HID) + (cc - 1)) * HW);
        plane += (yyc << 7);
        #pragma unroll
        for (int j = 0; j < 8; ++j) {
            int xx  = mbase + j * 16 + dx;
            int ok  = rowok & ((unsigned)xx < (unsigned)NW);
            int xxc = min(max(xx, 0), NW - 1);
            av[j]  = plane[xxc];
            af_[j] = ok ? 1.f : 0.f;
        }
    };
    auto loadB = [&](int kc) {
        const _Float16* src = Wp + (size_t)bn * KP + kc + bhs * 16;
        bv0 = *(const v4f*)src;
        bv1 = *(const v4f*)(src + 8);
    };
    auto storeAB = [&](int buf) {
        #pragma unroll
        for (int j = 0; j < 8; ++j)
            As[buf][(mbase + j * 16) * AS + ka] = (_Float16)(av[j] * af_[j]);
        _Float16* dst = Bs[buf] + bn * BS + bhs * 16;
        *(v4f*)dst       = bv0;
        *(v4f*)(dst + 8) = bv1;
    };

    loadA(0); loadB(0); storeAB(0);
    __syncthreads();

    for (int i = 0; i < NKC; ++i) {
        int cur = i & 1;
        // Issue next chunk's global loads FIRST; fence keeps them here so their
        // latency hides under the 8 WMMAs below.
        if (i + 1 < NKC) { loadA((i + 1) * 32); loadB((i + 1) * 32); }
        SCHED_FENCE();

        // Fragments + 8 dense WMMAs from the current LDS buffer.
        HalfFrag afr0, afr1, bfr[4];
        {
            const _Float16* ap0 = &As[cur][(mtA * 16 + lrow) * AS + lhi * 8];
            afr0.q[0] = *(const v4f*)ap0;
            afr0.q[1] = *(const v4f*)(ap0 + 16);
            const _Float16* ap1 = &As[cur][((mtA + 1) * 16 + lrow) * AS + lhi * 8];
            afr1.q[0] = *(const v4f*)ap1;
            afr1.q[1] = *(const v4f*)(ap1 + 16);
        }
        #pragma unroll
        for (int nt = 0; nt < 4; ++nt) {
            const _Float16* bp = &Bs[cur][(ngrp * 64 + nt * 16 + lrow) * BS + lhi * 8];
            bfr[nt].q[0] = *(const v4f*)bp;
            bfr[nt].q[1] = *(const v4f*)(bp + 16);
        }
        #pragma unroll
        for (int nt = 0; nt < 4; ++nt) {
            acc[nt]     = __builtin_amdgcn_wmma_f32_16x16x32_f16(
                false, afr0.v, false, bfr[nt].v, (short)0, acc[nt],     false, false);
            acc[4 + nt] = __builtin_amdgcn_wmma_f32_16x16x32_f16(
                false, afr1.v, false, bfr[nt].v, (short)0, acc[4 + nt], false, false);
        }

        // Stores (which wait on the global loads) must come AFTER the WMMAs.
        SCHED_FENCE();
        if (i + 1 < NKC) storeAB(cur ^ 1);
        __syncthreads();
    }

    // Epilogue: + bias, store G[pixel][channel] (lanes contiguous in channel)
    #pragma unroll
    for (int nt = 0; nt < 4; ++nt) {
        int o = ngrp * 64 + nt * 16 + lrow;
        float bo = bias[o];
        #pragma unroll
        for (int r = 0; r < 8; ++r) {
            int S0m = S0 + mtA * 16 + r + lhi * 8;
            G[(size_t)S0m * NG + o]        = acc[nt][r] + bo;
            G[(size_t)(S0m + 16) * NG + o] = acc[4 + nt][r] + bo;
        }
    }
}

// ---------------------------------------------------------------------------
// Pointwise ConvLSTM cell update with peephole Wc (3,HID,H,W)
// ---------------------------------------------------------------------------
__global__ void lstm_cell(const float* __restrict__ G,
                          float* __restrict__ h, float* __restrict__ c,
                          const float* __restrict__ Wc) {
    int i = blockIdx.x * blockDim.x + threadIdx.x;
    if (i >= NB * HID * HW) return;
    int pos = i & (HW - 1);
    int bu  = i >> 14;
    int u   = bu & 63, b = bu >> 6;
    size_t gb = ((size_t)(b * HW + pos)) * NG + u;
    float gi = G[gb], gf = G[gb + 64], gc = G[gb + 128], go = G[gb + 192];
    float cp = c[i];
    float wci = Wc[(0 * HID + u) * HW + pos];
    float wcf = Wc[(1 * HID + u) * HW + pos];
    float wco = Wc[(2 * HID + u) * HW + pos];
    float ig = 1.f / (1.f + expf(-(gi + cp * wci)));
    float fg = 1.f / (1.f + expf(-(gf + cp * wcf)));
    float cc = fg * cp + ig * tanhf(gc);
    float og = 1.f / (1.f + expf(-(go + cc * wco)));
    c[i] = cc;
    h[i] = og * tanhf(cc);
}

// ---------------------------------------------------------------------------
// uv = conv3x3(h, uv_w[2,HID,3,3]) + uv_b
// ---------------------------------------------------------------------------
__global__ __launch_bounds__(256)
void uv_conv(const float* __restrict__ h, const float* __restrict__ w,
             const float* __restrict__ bias, float* __restrict__ uv) {
    __shared__ float ws[2 * HID * 9];
    for (int j = threadIdx.x; j < 2 * HID * 9; j += blockDim.x) ws[j] = w[j];
    __syncthreads();
    int i = blockIdx.x * blockDim.x + threadIdx.x;     // NB*2*HW
    if (i >= NB * 2 * HW) return;
    int pos = i & (HW - 1);
    int bc  = i >> 14;
    int ch  = bc & 1, b = bc >> 1;
    int y = pos >> 7, x = pos & 127;
    float acc = bias[ch];
    const float* hb = h + (size_t)b * HID * HW;
    for (int c2 = 0; c2 < HID; ++c2) {
        const float* hc = hb + c2 * HW;
        const float* wc = ws + (ch * HID + c2) * 9;
        #pragma unroll
        for (int dy = -1; dy <= 1; ++dy) {
            int yy = y + dy;
            if (yy < 0 || yy >= NH) continue;
            #pragma unroll
            for (int dx = -1; dx <= 1; ++dx) {
                int xx = x + dx;
                if (xx < 0 || xx >= NW) continue;
                acc += hc[(yy << 7) + xx] * wc[(dy + 1) * 3 + (dx + 1)];
            }
        }
    }
    uv[i] = acc;
}

// ---------------------------------------------------------------------------
// Grid-sample uv at particle positions, advect, clip to [0,1]
// ---------------------------------------------------------------------------
__global__ void advect(const float* __restrict__ uv, float* __restrict__ xy) {
    int i = blockIdx.x * blockDim.x + threadIdx.x;    // NB*HW
    if (i >= NB * HW) return;
    int b = i >> 14, n = i & (HW - 1);
    float X = xy[(b * 2 + 0) * HW + n];
    float Y = xy[(b * 2 + 1) * HW + n];
    float px = X * NW - 0.5f, py = Y * NH - 0.5f;
    float x0f = floorf(px), y0f = floorf(py);
    float wx1 = px - x0f, wy1 = py - y0f;
    int x0 = (int)x0f, y0 = (int)y0f;
    const float* u0 = uv + (size_t)(b * 2 + 0) * HW;
    const float* u1 = uv + (size_t)(b * 2 + 1) * HW;
    float sx = 0.f, sy = 0.f;
    #pragma unroll
    for (int jy = 0; jy < 2; ++jy)
        #pragma unroll
        for (int jx = 0; jx < 2; ++jx) {
            int xx = x0 + jx, yy = y0 + jy;
            float wgt = (jx ? wx1 : 1.f - wx1) * (jy ? wy1 : 1.f - wy1);
            if (xx >= 0 && xx < NW && yy >= 0 && yy < NH) {
                int p = (yy << 7) + xx;
                sx += u0[p] * wgt;
                sy += u1[p] * wgt;
            }
        }
    xy[(b * 2 + 0) * HW + n] = fminf(fmaxf(X + sx, 0.f), 1.f);
    xy[(b * 2 + 1) * HW + n] = fminf(fmaxf(Y + sy, 0.f), 1.f);
}

__global__ void zero2(float* __restrict__ a, float* __restrict__ b, int n) {
    int i = blockIdx.x * blockDim.x + threadIdx.x;
    if (i < n) { a[i] = 0.f; b[i] = 0.f; }
}

// ---------------------------------------------------------------------------
// Bilinear scatter of R_pc = x[:, T-1] at xy into acc/wacc
// ---------------------------------------------------------------------------
__global__ void scatter(const float* __restrict__ x, const float* __restrict__ xy,
                        float* __restrict__ acc, float* __restrict__ wacc) {
    int i = blockIdx.x * blockDim.x + threadIdx.x;    // NB*HW
    if (i >= NB * HW) return;
    int b = i >> 14, n = i & (HW - 1);
    float X = xy[(b * 2 + 0) * HW + n];
    float Y = xy[(b * 2 + 1) * HW + n];
    float px = X * (NW - 1), py = Y * (NH - 1);
    float x0f = floorf(px), y0f = floorf(py);
    float wx1 = px - x0f, wy1 = py - y0f;
    int x0 = min(max((int)x0f, 0), NW - 1);
    int y0 = min(max((int)y0f, 0), NH - 1);
    int x1 = min(x0 + 1, NW - 1);
    int y1 = min(y0 + 1, NH - 1);
    float r = x[((size_t)(b * NT + (NT - 1))) * HW + n];
    float w00 = (1.f - wx1) * (1.f - wy1), w10 = wx1 * (1.f - wy1);
    float w01 = (1.f - wx1) * wy1,         w11 = wx1 * wy1;
    int base = b * HW;
    atomicAdd(&acc[base + y0 * NW + x0], r * w00); atomicAdd(&wacc[base + y0 * NW + x0], w00);
    atomicAdd(&acc[base + y0 * NW + x1], r * w10); atomicAdd(&wacc[base + y0 * NW + x1], w10);
    atomicAdd(&acc[base + y1 * NW + x0], r * w01); atomicAdd(&wacc[base + y1 * NW + x0], w01);
    atomicAdd(&acc[base + y1 * NW + x1], r * w11); atomicAdd(&wacc[base + y1 * NW + x1], w11);
}

__global__ void normalize(const float* __restrict__ acc, const float* __restrict__ wacc,
                          float* __restrict__ rg, float* __restrict__ out, int t) {
    int i = blockIdx.x * blockDim.x + threadIdx.x;    // NB*HW
    if (i >= NB * HW) return;
    int b = i >> 14, n = i & (HW - 1);
    float v = acc[i] / fmaxf(wacc[i], 1e-8f);
    rg[i] = v;
    out[((size_t)b * NT + t) * HW + n] = v;
}

// ---------------------------------------------------------------------------
// Init: h=c=0, xy = normalized meshgrid, rg = x[:, T-1]
// ---------------------------------------------------------------------------
__global__ void init_state(float* __restrict__ h, float* __restrict__ c,
                           float* __restrict__ xy, float* __restrict__ rg,
                           const float* __restrict__ x) {
    int i = blockIdx.x * blockDim.x + threadIdx.x;
    if (i < NB * HID * HW) { h[i] = 0.f; c[i] = 0.f; }
    if (i < NB * 2 * HW) {
        int ch = (i >> 14) & 1;
        int n  = i & (HW - 1);
        xy[i] = ch == 0 ? (float)(n & 127) * (1.0f / (NW - 1))
                        : (float)(n >> 7)  * (1.0f / (NH - 1));
    }
    if (i < NB * HW) {
        int b = i >> 14, n = i & (HW - 1);
        rg[i] = x[((size_t)(b * NT + (NT - 1))) * HW + n];
    }
}

extern "C" void kernel_launch(void* const* d_in, const int* in_sizes, int n_in,
                              void* d_out, int out_size, void* d_ws, size_t ws_size,
                              hipStream_t stream) {
    const float* x        = (const float*)d_in[0];
    const float* enc_Wx   = (const float*)d_in[1];
    const float* enc_Wxb  = (const float*)d_in[2];
    const float* enc_Wh   = (const float*)d_in[3];
    const float* enc_Wc   = (const float*)d_in[4];
    const float* pred_Wx  = (const float*)d_in[5];
    const float* pred_Wxb = (const float*)d_in[6];
    const float* pred_Wh  = (const float*)d_in[7];
    const float* pred_Wc  = (const float*)d_in[8];
    const float* uv_w     = (const float*)d_in[9];
    const float* uv_b     = (const float*)d_in[10];
    float* out = (float*)d_out;

    char* ws = (char*)d_ws;
    auto take = [&](size_t bytes) {
        char* p = ws;
        ws += (bytes + 255) & ~(size_t)255;
        return p;
    };
    float*     hbuf  = (float*)take((size_t)NB * HID * HW * 4);
    float*     cbuf  = (float*)take((size_t)NB * HID * HW * 4);
    float*     G     = (float*)take((size_t)NB * HW * NG * 4);
    _Float16*  WpE   = (_Float16*)take((size_t)NG * KP * 2);
    _Float16*  WpP   = (_Float16*)take((size_t)NG * KP * 2);
    float*     uvb   = (float*)take((size_t)NB * 2 * HW * 4);
    float*     xyb   = (float*)take((size_t)NB * 2 * HW * 4);
    float*     rgb   = (float*)take((size_t)NB * HW * 4);
    float*     accb  = (float*)take((size_t)NB * HW * 4);
    float*     waccb = (float*)take((size_t)NB * HW * 4);

    const int tpb = 256;
    pack_weights<<<(NG * KP + tpb - 1) / tpb, tpb, 0, stream>>>(enc_Wx, enc_Wh, WpE);
    pack_weights<<<(NG * KP + tpb - 1) / tpb, tpb, 0, stream>>>(pred_Wx, pred_Wh, WpP);
    init_state<<<(NB * HID * HW + tpb - 1) / tpb, tpb, 0, stream>>>(hbuf, cbuf, xyb, rgb, x);

    const int ggrid = NB * HW / 128;   // 1024 blocks (one image row each)
    int nCell = NB * HID * HW;

    // Encoder: 12 ConvLSTM steps on x[:, t]
    for (int t = 0; t < NT; ++t) {
        gates_wmma<<<ggrid, 512, 0, stream>>>(x + (size_t)t * HW, NT * HW,
                                              hbuf, WpE, enc_Wxb, G);
        lstm_cell<<<(nCell + tpb - 1) / tpb, tpb, 0, stream>>>(G, hbuf, cbuf, enc_Wc);
    }
    // Predictor: 12 steps
    for (int t = 0; t < NT; ++t) {
        gates_wmma<<<ggrid, 512, 0, stream>>>(rgb, HW, hbuf, WpP, pred_Wxb, G);
        lstm_cell<<<(nCell + tpb - 1) / tpb, tpb, 0, stream>>>(G, hbuf, cbuf, pred_Wc);
        uv_conv<<<(NB * 2 * HW + tpb - 1) / tpb, tpb, 0, stream>>>(hbuf, uv_w, uv_b, uvb);
        advect<<<(NB * HW + tpb - 1) / tpb, tpb, 0, stream>>>(uvb, xyb);
        zero2<<<(NB * HW + tpb - 1) / tpb, tpb, 0, stream>>>(accb, waccb, NB * HW);
        scatter<<<(NB * HW + tpb - 1) / tpb, tpb, 0, stream>>>(x, xyb, accb, waccb);
        normalize<<<(NB * HW + tpb - 1) / tpb, tpb, 0, stream>>>(accb, waccb, rgb, out, t);
    }
}
